// ExplainableCrossAttention_69887707840833
// MI455X (gfx1250) — compile-verified, matrix operand order checked by reading, not run on hardware
//
#include <hip/hip_runtime.h>
#include <hip/hip_bf16.h>

#define M_  128
#define L_  256
#define H_  768
#define NH_ 12
#define HD_ 64
#define N_  (M_ * L_)
#define KC_ 384   // K-chunk staged to LDS per TDM transfer (64 x 384 f16 = 48KB)

typedef _Float16 half_t;
typedef __attribute__((ext_vector_type(16))) _Float16 v16h;
typedef __attribute__((ext_vector_type(8)))  _Float16 v8h;
typedef __attribute__((ext_vector_type(4)))  _Float16 v4h;
typedef __attribute__((ext_vector_type(8)))  float    v8f;
typedef unsigned int u32x4 __attribute__((ext_vector_type(4)));
typedef int          i32x4 __attribute__((ext_vector_type(4)));
typedef int          i32x8 __attribute__((ext_vector_type(8)));

union Frag { v16h v; v8h h[2]; };

// CDNA5 16-bit A/B fragment: lanes 0-15 hold rows 0-15 with K = {0..7, 16..23},
// lanes 16-31 hold the same rows with K = {8..15, 24..31}. Two b128 loads/lane.
__device__ __forceinline__ Frag load_frag(const half_t* base, int ld, int lane) {
  Frag f;
  const half_t* p = base + (size_t)(lane & 15) * ld + ((lane >> 4) << 3);
  f.h[0] = *(const v8h*)p;
  f.h[1] = *(const v8h*)(p + 16);
  return f;
}

__device__ __forceinline__ v8f wmma_f16(Frag a, Frag b, v8f c) {
  return __builtin_amdgcn_wmma_f32_16x16x32_f16(false, a.v, false, b.v,
                                                (short)0, c, false, false);
}

// TDM: DMA a 64-row x KC_-col f16 tile of W (row stride H_) into LDS.
// D# built per CDNA5 ISA. Issued by one wave.
__device__ __forceinline__ void tdm_stage_wtile(const half_t* gsrc,
                                                unsigned lds_base) {
  unsigned long long ga = (unsigned long long)(uintptr_t)gsrc;
  u32x4 g0 = { 1u,                                     // count=1, is_restore=0
               lds_base,                               // lds_addr (bytes)
               (unsigned)(ga & 0xffffffffull),         // global_addr[31:0]
               (unsigned)((ga >> 32) & 0x01ffffffu) | (2u << 30) };  // type=2
  const unsigned td0 = KC_, td1 = 64, tile0 = KC_, tile1 = 64, st0 = H_;
  i32x8 g1 = { (int)(1u << 16),                        // data_size = 2 bytes
               (int)(td0 << 16),                       // tensor_dim0[15:0]
               (int)((td0 >> 16) | (td1 << 16)),       // td0 hi | tensor_dim1 lo
               (int)((td1 >> 16) | (tile0 << 16)),     // td1 hi | tile_dim0
               (int)tile1,                             // tile_dim1 (tile_dim2=0)
               (int)st0,                               // tensor_dim0_stride lo32
               0, 0 };
  i32x4 g2 = { 0, 0, 0, 0 };
  i32x4 g3 = { 0, 0, 0, 0 };
#if defined(__clang_major__) && (__clang_major__ >= 23)
  i32x8 g4 = { 0, 0, 0, 0, 0, 0, 0, 0 };
  __builtin_amdgcn_tensor_load_to_lds(g0, g1, g2, g3, g4, 0);
#else
  __builtin_amdgcn_tensor_load_to_lds(g0, g1, g2, g3, 0);
#endif
  __builtin_amdgcn_s_wait_tensorcnt(0);
}

// ---------------------------------------------------------------- converts
__global__ __launch_bounds__(256) void conv_x_kernel(
    const float* __restrict__ atom, const int* __restrict__ a_starts,
    const int* __restrict__ a_sizes, half_t* __restrict__ xh) {
  int idx = blockIdx.x * blockDim.x + threadIdx.x;     // one float4
  int g   = idx / (H_ / 4);
  int c4  = idx % (H_ / 4);
  int m   = g >> 8;
  int pos = g & (L_ - 1);
  float4 v = make_float4(0.f, 0.f, 0.f, 0.f);
  if (pos < a_sizes[m]) {
    const float4* src = (const float4*)(atom + (size_t)(a_starts[m] + pos) * H_);
    v = src[c4];
  }
  v4h o = { (half_t)v.x, (half_t)v.y, (half_t)v.z, (half_t)v.w };
  *(v4h*)(xh + (size_t)g * H_ + c4 * 4) = o;
}

__global__ __launch_bounds__(256) void conv_w_kernel(
    const float* __restrict__ W, half_t* __restrict__ Wh) {
  int idx = blockIdx.x * blockDim.x + threadIdx.x;     // one float4
  float4 v = ((const float4*)W)[idx];
  v4h o = { (half_t)v.x, (half_t)v.y, (half_t)v.z, (half_t)v.w };
  *(v4h*)(Wh + (size_t)idx * 4) = o;
}

// ---------------------------------------------------------------- GEMM core
// acc[4] += X[row0:+16, :] * W[n0:+64, :]^T ; W tile TDM-staged through LDS,
// A fragments double-buffered from global.
__device__ __forceinline__ void gemm_core(const half_t* __restrict__ Xh,
                                          const half_t* __restrict__ Wh,
                                          half_t* wtile, int row0, int n0,
                                          int lane, v8f acc[4]) {
  const half_t* arow = Xh + (size_t)row0 * H_;
  for (int kc = 0; kc < H_; kc += KC_) {
    if (threadIdx.x < 32)
      tdm_stage_wtile(Wh + (size_t)n0 * H_ + kc,
                      (unsigned)(uintptr_t)(void*)wtile);
    __syncthreads();
    Frag a0 = load_frag(arow + kc, H_, lane);
#pragma unroll 1
    for (int k0 = 0; k0 < KC_ - 32; k0 += 32) {
      Frag a1 = load_frag(arow + kc + k0 + 32, H_, lane);   // prefetch next K
#pragma unroll
      for (int t = 0; t < 4; ++t) {
        Frag b = load_frag(wtile + (size_t)(t * 16) * KC_ + k0, KC_, lane);
        acc[t] = wmma_f16(a0, b, acc[t]);
      }
      a0 = a1;
    }
#pragma unroll
    for (int t = 0; t < 4; ++t) {                            // peeled last step
      Frag b = load_frag(wtile + (size_t)(t * 16) * KC_ + (KC_ - 32), KC_, lane);
      acc[t] = wmma_f16(a0, b, acc[t]);
    }
    __syncthreads();   // protect wtile before next chunk overwrites it
  }
}

// ---------------------------------------------------------------- QKV GEMM
__global__ __launch_bounds__(256) void qkv_gemm_kernel(
    const half_t* __restrict__ Xh, const half_t* __restrict__ Wh,
    const float* __restrict__ bias, half_t* __restrict__ dst, int vtransposed) {
  __shared__ __attribute__((aligned(16))) half_t wtile[64 * KC_];
  int lane = threadIdx.x & 31, wave = threadIdx.x >> 5;
  int rb = blockIdx.x % (N_ / 128);
  int cb = blockIdx.x / (N_ / 128);
  int row0 = rb * 128 + wave * 16;
  int n0   = cb * 64;
  v8f acc[4] = {};
  gemm_core(Xh, Wh, wtile, row0, n0, lane, acc);

  int hl = lane >> 4, nn = lane & 15;
#pragma unroll
  for (int t = 0; t < 4; ++t) {
    int o = n0 + t * 16 + nn;
    float bb = bias[o];
    int hh = o >> 6, dd = o & 63;
#pragma unroll
    for (int i = 0; i < 8; ++i) {
      int g = row0 + i + 8 * hl;
      int mm = g >> 8, ll = g & (L_ - 1);
      size_t off = vtransposed
          ? ((size_t)(mm * NH_ + hh) * HD_ + dd) * L_ + ll    // V: [m,h,d,l]
          : ((size_t)(mm * NH_ + hh) * L_ + ll) * HD_ + dd;   // Q,K: [m,h,l,d]
      dst[off] = (half_t)(acc[t][i] + bb);
    }
  }
}

// ---------------------------------------------------------------- attention
// One block per (m, h); each wave owns 32 queries; flash-style online softmax.
__global__ __launch_bounds__(256) void attn_kernel(
    const half_t* __restrict__ qh, const half_t* __restrict__ kh,
    const half_t* __restrict__ vt, const int* __restrict__ a_sizes,
    half_t* __restrict__ ctxh) {
  __shared__ __attribute__((aligned(16))) half_t plds[8][16][32];
  int lane = threadIdx.x & 31, wave = threadIdx.x >> 5;
  int m = blockIdx.x / NH_, h = blockIdx.x % NH_;
  int size = a_sizes[m];
  const half_t* qb = qh + (size_t)(m * NH_ + h) * L_ * HD_;
  const half_t* kb = kh + (size_t)(m * NH_ + h) * L_ * HD_;
  const half_t* vb = vt + (size_t)(m * NH_ + h) * HD_ * L_;
  int q0 = wave * 32;
  int hl = lane >> 4, nn = lane & 15;

  Frag qf[2][2];
#pragma unroll
  for (int t = 0; t < 2; ++t)
#pragma unroll
    for (int kf = 0; kf < 2; ++kf)
      qf[t][kf] = load_frag(qb + (size_t)(q0 + t * 16) * HD_ + kf * 32, HD_, lane);

  v8f acc[2][4] = {};
  float rm[2][8], rs[2][8];
#pragma unroll
  for (int t = 0; t < 2; ++t)
#pragma unroll
    for (int i = 0; i < 8; ++i) { rm[t][i] = -3.0e38f; rs[t][i] = 0.f; }

  const float scale = 0.125f;   // HD^-0.5
  for (int j0 = 0; j0 < L_; j0 += 32) {
    Frag kfr[2][2], vfr[4];
#pragma unroll
    for (int s = 0; s < 2; ++s)
#pragma unroll
      for (int kf = 0; kf < 2; ++kf)
        kfr[s][kf] = load_frag(kb + (size_t)(j0 + s * 16) * HD_ + kf * 32, HD_, lane);
#pragma unroll
    for (int d = 0; d < 4; ++d)
      vfr[d] = load_frag(vb + (size_t)(d * 16) * L_ + j0, L_, lane);   // transposed V

    bool ok0 = (j0 + nn) < size;
    bool ok1 = (j0 + 16 + nn) < size;
#pragma unroll
    for (int t = 0; t < 2; ++t) {
      v8f s0 = {}, s1 = {};
      s0 = wmma_f16(qf[t][0], kfr[0][0], s0);
      s0 = wmma_f16(qf[t][1], kfr[0][1], s0);
      s1 = wmma_f16(qf[t][0], kfr[1][0], s1);
      s1 = wmma_f16(qf[t][1], kfr[1][1], s1);
#pragma unroll
      for (int i = 0; i < 8; ++i) {
        float a = ok0 ? s0[i] * scale : -1e9f;
        float b = ok1 ? s1[i] * scale : -1e9f;
        float mx = fmaxf(a, b);
#pragma unroll
        for (int w = 1; w < 16; w <<= 1) mx = fmaxf(mx, __shfl_xor(mx, w, 32));
        float nm = fmaxf(rm[t][i], mx);
        float alpha = __expf(rm[t][i] - nm);
        rm[t][i] = nm;
        float e0 = __expf(a - nm);
        float e1 = __expf(b - nm);
        float ls = e0 + e1;
#pragma unroll
        for (int w = 1; w < 16; w <<= 1) ls += __shfl_xor(ls, w, 32);
        rs[t][i] = rs[t][i] * alpha + ls;
#pragma unroll
        for (int d = 0; d < 4; ++d) acc[t][d][i] *= alpha;
        int r = i + 8 * hl;               // re-layout C-frag -> A-frag via LDS
        plds[wave][r][nn]      = (half_t)e0;
        plds[wave][r][16 + nn] = (half_t)e1;
      }
      Frag pf = load_frag(&plds[wave][0][0], 32, lane);  // wave-private, in-order DS
#pragma unroll
      for (int d = 0; d < 4; ++d)
        acc[t][d] = wmma_f16(pf, vfr[d], acc[t][d]);
    }
  }
#pragma unroll
  for (int t = 0; t < 2; ++t) {
    float inv[8];
#pragma unroll
    for (int i = 0; i < 8; ++i) inv[i] = 1.f / rs[t][i];
#pragma unroll
    for (int d = 0; d < 4; ++d)
#pragma unroll
      for (int i = 0; i < 8; ++i) {
        float val = acc[t][d][i] * inv[i];
        int row = q0 + t * 16 + i + 8 * hl;
        int col = h * HD_ + d * 16 + nn;
        ctxh[(size_t)(m * L_ + row) * H_ + col] = (half_t)val;
      }
  }
}

// ------------------------------------------------- output proj + residual
__global__ __launch_bounds__(256) void proj_gemm_kernel(
    const half_t* __restrict__ Ch, const half_t* __restrict__ Woh,
    const float* __restrict__ bo, const float* __restrict__ atom,
    const int* __restrict__ a_starts, const int* __restrict__ a_sizes,
    float* __restrict__ ypre) {
  __shared__ __attribute__((aligned(16))) half_t wtile[64 * KC_];
  int lane = threadIdx.x & 31, wave = threadIdx.x >> 5;
  int rb = blockIdx.x % (N_ / 128);
  int cb = blockIdx.x / (N_ / 128);
  int row0 = rb * 128 + wave * 16;
  int n0   = cb * 64;
  v8f acc[4] = {};
  gemm_core(Ch, Woh, wtile, row0, n0, lane, acc);

  int hl = lane >> 4, nn = lane & 15;
  int    gi[8]; size_t xs[8]; bool vld[8];
#pragma unroll
  for (int i = 0; i < 8; ++i) {
    int g = row0 + i + 8 * hl;
    gi[i] = g;
    int mm = g >> 8, pos = g & (L_ - 1);
    vld[i] = pos < a_sizes[mm];
    xs[i]  = (size_t)(a_starts[mm] + pos) * H_;
  }
#pragma unroll
  for (int t = 0; t < 4; ++t) {
    int o = n0 + t * 16 + nn;
    float bb = bo[o];
#pragma unroll
    for (int i = 0; i < 8; ++i) {
      float xval = vld[i] ? atom[xs[i] + o] : 0.f;    // residual (masked x)
      ypre[(size_t)gi[i] * H_ + o] = acc[t][i] + bb + xval;
    }
  }
}

// ---------------------------------------------------------------- layernorm
__global__ __launch_bounds__(256) void ln_kernel(
    const float* __restrict__ ypre, const float* __restrict__ gamma,
    const float* __restrict__ beta, const int* __restrict__ a_starts,
    const int* __restrict__ a_sizes, float* __restrict__ out) {
  int lane = threadIdx.x & 31, wave = threadIdx.x >> 5;
  int g = blockIdx.x * 8 + wave;
  int mm = g >> 8, pos = g & (L_ - 1);
  const float* row = ypre + (size_t)g * H_;
  float s = 0.f, s2 = 0.f;
  for (int c = lane; c < H_; c += 32) { float v = row[c]; s += v; s2 += v * v; }
#pragma unroll
  for (int w = 1; w < 32; w <<= 1) {
    s  += __shfl_xor(s,  w, 32);
    s2 += __shfl_xor(s2, w, 32);
  }
  float mu   = s * (1.f / H_);
  float var  = s2 * (1.f / H_) - mu * mu;
  float rstd = rsqrtf(var + 1e-5f);
  float msk  = (pos < a_sizes[mm]) ? 1.f : 0.f;
  float* drow = out + (size_t)(a_starts[mm] + pos) * H_;
  for (int c = lane; c < H_; c += 32)
    drow[c] = ((row[c] - mu) * rstd * gamma[c] + beta[c]) * msk;
}

// ---------------------------------------------------------------- launch
extern "C" void kernel_launch(void* const* d_in, const int* in_sizes, int n_in,
                              void* d_out, int out_size, void* d_ws, size_t ws_size,
                              hipStream_t stream) {
  const float* atom  = (const float*)d_in[0];
  const float* Wq    = (const float*)d_in[1];
  const float* bq    = (const float*)d_in[2];
  const float* Wk    = (const float*)d_in[3];
  const float* bk    = (const float*)d_in[4];
  const float* Wv    = (const float*)d_in[5];
  const float* bv    = (const float*)d_in[6];
  const float* Wo    = (const float*)d_in[7];
  const float* bo    = (const float*)d_in[8];
  const float* gamma = (const float*)d_in[9];
  const float* beta  = (const float*)d_in[10];
  const int* a_starts = (const int*)d_in[11];
  const int* a_sizes  = (const int*)d_in[12];

  char* ws = (char*)d_ws;
  const size_t NBh = (size_t)N_ * H_ * sizeof(half_t);
  half_t* xh  = (half_t*)ws;               // later reused as ctx
  half_t* qh  = (half_t*)(ws + NBh);
  half_t* kh  = (half_t*)(ws + 2 * NBh);
  half_t* vh  = (half_t*)(ws + 3 * NBh);   // transposed V
  half_t* Wqh = (half_t*)(ws + 4 * NBh);
  half_t* Wkh = Wqh + (size_t)H_ * H_;
  half_t* Wvh = Wkh + (size_t)H_ * H_;
  half_t* Woh = Wvh + (size_t)H_ * H_;
  half_t* ctxh = xh;                        // alias: xh dead after QKV GEMMs
  float*  ypre = (float*)(ws + NBh);        // alias: qh+kh dead after attention

  hipMemsetAsync(d_out, 0, (size_t)out_size * sizeof(float), stream);

  conv_x_kernel<<<(N_ * (H_ / 4)) / 256, 256, 0, stream>>>(atom, a_starts, a_sizes, xh);
  int wgrid = (H_ * H_ / 4) / 256;
  conv_w_kernel<<<wgrid, 256, 0, stream>>>(Wq, Wqh);
  conv_w_kernel<<<wgrid, 256, 0, stream>>>(Wk, Wkh);
  conv_w_kernel<<<wgrid, 256, 0, stream>>>(Wv, Wvh);
  conv_w_kernel<<<wgrid, 256, 0, stream>>>(Wo, Woh);

  dim3 ggrid((N_ / 128) * (H_ / 64));
  qkv_gemm_kernel<<<ggrid, 256, 0, stream>>>(xh, Wqh, bq, qh, 0);
  qkv_gemm_kernel<<<ggrid, 256, 0, stream>>>(xh, Wkh, bk, kh, 0);
  qkv_gemm_kernel<<<ggrid, 256, 0, stream>>>(xh, Wvh, bv, vh, 1);

  attn_kernel<<<M_ * NH_, 256, 0, stream>>>(qh, kh, vh, a_sizes, ctxh);

  proj_gemm_kernel<<<ggrid, 256, 0, stream>>>(ctxh, Woh, bo, atom, a_starts,
                                              a_sizes, ypre);

  ln_kernel<<<N_ / 8, 256, 0, stream>>>(ypre, gamma, beta, a_starts, a_sizes,
                                        (float*)d_out);
}
